// AMD_MIL_81862076662080
// MI455X (gfx1250) — matrix operand order repaired and checked by reading, not compiled, not run on hardware
//
#include <hip/hip_runtime.h>
#include <hip/hip_bf16.h>
#include <math.h>

// ---------------- constants (shapes fixed by the reference) ----------------
static constexpr int BATCH = 2;
static constexpr int NTOK0 = 10000;     // patches
static constexpr int GRID  = 100;       // H = W = 100
static constexpr int TOK   = 10001;     // +cls
static constexpr int DI    = 1024;
static constexpr int DH    = 512;
static constexpr int Q3    = 3 * DH;    // 1536
static constexpr int NH    = 8;
static constexpr int HD    = 64;
static constexpr int AG    = 256;

typedef __attribute__((ext_vector_type(16))) _Float16 v16h;
typedef __attribute__((ext_vector_type(8)))  _Float16 v8h;
typedef __attribute__((ext_vector_type(4)))  _Float16 v4h;
typedef __attribute__((ext_vector_type(8)))  float    v8f;

// =====================  generalized WMMA GEMM  =====================
// C[z] = relu?( alpha * A[z] x B[z] + bias )      (f32 in, f16 WMMA, f32 out)
// z = zb * zH + zh ; per-operand offsets: base + zb*sZb + zh*sZh
struct GemmP {
  const float* A; long long aZb, aZh; int lda;
  const float* B; long long bZb, bZh; int ldb; int transB;
  float*       C; long long cZb, cZh; int ldc;
  const float* bias;
  int M, N, K, zH, relu;
  int aVec, bVec;            // lda/ldb % 4 == 0 -> float4 staging allowed
  float alpha;
};

#define TM 128
#define TN 128
#define TK 32
#define FRAG_STRIDE 24   // halves per lane slot-row (48B: 16B aligned, 2-way conflicts)

__global__ __launch_bounds__(256) void k_gemm(GemmP p) {
  // LDS tiles stored directly in WMMA fragment layout:
  //   As[mTile 0..7][lane 0..31][slot 0..15]   (A 16x32 f16 map)
  //   Bs[nTile 0..7][lane 0..31][slot 0..15]   (B 32x16 f16 map)
  __shared__ _Float16 As[8][32][FRAG_STRIDE];
  __shared__ _Float16 Bs[8][32][FRAG_STRIDE];

  const int tid   = threadIdx.x;
  const int wave  = tid >> 5;
  const int lane  = tid & 31;
  const int l15   = lane & 15;
  const int half  = lane >> 4;
  const int waveM = wave >> 1;       // 0..3 -> rows  waveM*32
  const int waveN = wave & 1;        // 0..1 -> cols  waveN*64

  const int z  = blockIdx.z;
  const int zb = z / p.zH;
  const int zh = z - zb * p.zH;
  const float* Ab = p.A + zb * p.aZb + zh * p.aZh;
  const float* Bb = p.B + zb * p.bZb + zh * p.bZh;
  float*       Cb = p.C + zb * p.cZb + zh * p.cZh;

  const int m0 = blockIdx.x * TM;
  const int n0 = blockIdx.y * TN;
  const int Mm1 = p.M - 1, Km1 = p.K - 1, Nm1 = p.N - 1;

  v8f acc[2][4];
  #pragma unroll
  for (int i = 0; i < 2; ++i)
    #pragma unroll
    for (int j = 0; j < 4; ++j)
      #pragma unroll
      for (int r = 0; r < 8; ++r) acc[i][j][r] = 0.f;

  for (int k0 = 0; k0 < p.K; k0 += TK) {
    const bool ktail = (k0 + TK > p.K);

    // ---------------- stage A (TM x TK) ----------------
    if (p.aVec && !ktail) {
      #pragma unroll
      for (int t = 0; t < 4; ++t) {
        int i  = tid + t * 256;            // 0..1023 float4s
        int r  = i >> 3;                   // row in tile
        int kq = (i & 7) << 2;             // k quad base
        int gm = m0 + r;
        int gmc = gm < p.M ? gm : Mm1;     // clamped, always valid
        const float4 v = *(const float4*)(Ab + (long long)gmc * p.lda + k0 + kq);
        float msk = (gm < p.M) ? 1.f : 0.f;
        v4h h;
        h[0] = (_Float16)(v.x * msk); h[1] = (_Float16)(v.y * msk);
        h[2] = (_Float16)(v.z * msk); h[3] = (_Float16)(v.w * msk);
        int ln = (r & 15) + ((kq >> 4) << 4);
        int j  = kq & 15;
        *(v4h*)&As[r >> 4][ln][j] = h;
      }
      // prefetch next A tile rows (global_prefetch)
      if (k0 + 2 * TK <= p.K && tid < 128) {
        int gm = m0 + tid;
        int gmc = gm < p.M ? gm : Mm1;
        __builtin_prefetch(Ab + (long long)gmc * p.lda + k0 + TK, 0, 1);
      }
    } else {
      #pragma unroll
      for (int t = 0; t < 16; ++t) {
        int i = tid + t * 256;             // 0..4095 elements
        int r = i >> 5, c = i & 31;
        int gm = m0 + r, gk = k0 + c;
        int gmc = gm < p.M ? gm : Mm1;
        int gkc = gk < p.K ? gk : Km1;
        float v = Ab[(long long)gmc * p.lda + gkc];
        v = (gm < p.M && gk < p.K) ? v : 0.f;
        As[r >> 4][(r & 15) + ((c >> 4) << 4)][c & 15] =
            (_Float16)v;
      }
    }

    // ---------------- stage B (TK x TN) ----------------
    if (p.bVec && !ktail) {
      if (p.transB) {
        // contiguous along K in memory -> float4 load + ds_store_b64
        #pragma unroll
        for (int t = 0; t < 4; ++t) {
          int i  = tid + t * 256;
          int n  = i >> 3;                 // col in tile
          int kq = (i & 7) << 2;
          int gn = n0 + n;
          int gnc = gn < p.N ? gn : Nm1;
          const float4 v = *(const float4*)(Bb + (long long)gnc * p.ldb + k0 + kq);
          float msk = (gn < p.N) ? 1.f : 0.f;
          v4h h;
          h[0] = (_Float16)(v.x * msk); h[1] = (_Float16)(v.y * msk);
          h[2] = (_Float16)(v.z * msk); h[3] = (_Float16)(v.w * msk);
          *(v4h*)&Bs[n >> 4][(n & 15) + ((kq >> 4) << 4)][kq & 15] = h;
        }
      } else {
        // contiguous along N in memory -> float4 load + 4 ds_store_b16 scatter
        #pragma unroll
        for (int t = 0; t < 4; ++t) {
          int i  = tid + t * 256;
          int k  = i >> 5;                 // k row (0..31)
          int nq = (i & 31) << 2;          // n quad base
          int gn = n0 + nq;
          int gnc = gn < p.N ? gn : (p.N - 4);   // N % 4 == 0 for all non-trans uses
          const float4 v = *(const float4*)(Bb + (long long)(k0 + k) * p.ldb + gnc);
          float msk = (gn < p.N) ? 1.f : 0.f;
          float vv[4] = {v.x * msk, v.y * msk, v.z * msk, v.w * msk};
          int hi = (k >> 4) << 4, j = k & 15;
          #pragma unroll
          for (int e = 0; e < 4; ++e) {
            int n = nq + e;
            Bs[n >> 4][(n & 15) + hi][j] = (_Float16)vv[e];
          }
        }
      }
    } else {
      #pragma unroll
      for (int t = 0; t < 16; ++t) {
        int i = tid + t * 256;
        int r = i >> 7, c = i & 127;       // r = k, c = n
        int gk = k0 + r, gn = n0 + c;
        int gkc = gk < p.K ? gk : Km1;
        int gnc = gn < p.N ? gn : Nm1;
        float v = p.transB ? Bb[(long long)gnc * p.ldb + gkc]
                           : Bb[(long long)gkc * p.ldb + gnc];
        v = (gk < p.K && gn < p.N) ? v : 0.f;
        Bs[c >> 4][(c & 15) + ((r >> 4) << 4)][r & 15] = (_Float16)v;
      }
    }
    __syncthreads();

    // ---------------- fragments: contiguous 16B LDS vector loads ----------------
    v16h af[2];
    #pragma unroll
    for (int tm = 0; tm < 2; ++tm) {
      const _Float16* src = &As[waveM * 2 + tm][lane][0];
      v8h lo = *(const v8h*)(src);
      v8h hi = *(const v8h*)(src + 8);
      af[tm] = __builtin_shufflevector(lo, hi, 0, 1, 2, 3, 4, 5, 6, 7,
                                       8, 9, 10, 11, 12, 13, 14, 15);
    }
    v16h bfr[4];
    #pragma unroll
    for (int tn = 0; tn < 4; ++tn) {
      const _Float16* src = &Bs[waveN * 4 + tn][lane][0];
      v8h lo = *(const v8h*)(src);
      v8h hi = *(const v8h*)(src + 8);
      bfr[tn] = __builtin_shufflevector(lo, hi, 0, 1, 2, 3, 4, 5, 6, 7,
                                        8, 9, 10, 11, 12, 13, 14, 15);
    }
    #pragma unroll
    for (int tm = 0; tm < 2; ++tm)
      #pragma unroll
      for (int tn = 0; tn < 4; ++tn)
        acc[tm][tn] = __builtin_amdgcn_wmma_f32_16x16x32_f16(
            false, af[tm], false, bfr[tn], (short)0, acc[tm][tn], false, false);
    __syncthreads();
  }

  // ---- store C (VGPR r: M = r + 8*half, N = lane&15) ----
  #pragma unroll
  for (int tm = 0; tm < 2; ++tm) {
    #pragma unroll
    for (int tn = 0; tn < 4; ++tn) {
      int n = n0 + waveN * 64 + tn * 16 + l15;
      if (n >= p.N) continue;
      float bia = p.bias ? p.bias[n] : 0.f;
      #pragma unroll
      for (int r = 0; r < 8; ++r) {
        int m = m0 + waveM * 32 + tm * 16 + r + half * 8;
        if (m < p.M) {
          float v = acc[tm][tn][r] * p.alpha + bia;
          if (p.relu) v = fmaxf(v, 0.f);
          Cb[(long long)m * p.ldc + n] = v;
        }
      }
    }
  }
}

// =====================  elementwise / reduction kernels  =====================

__global__ __launch_bounds__(128) void k_layernorm(const float* __restrict__ x,
                                                   float* __restrict__ y,
                                                   const float* __restrict__ g,
                                                   const float* __restrict__ be) {
  __shared__ float rs[128], rq[128];
  size_t row = blockIdx.x;
  const float4* p = (const float4*)(x + row * DH);
  float4* o = (float4*)(y + row * DH);
  int tid = threadIdx.x;
  float4 v = p[tid];
  float vv[4] = {v.x, v.y, v.z, v.w};
  float s = vv[0] + vv[1] + vv[2] + vv[3];
  float ss = vv[0]*vv[0] + vv[1]*vv[1] + vv[2]*vv[2] + vv[3]*vv[3];
  rs[tid] = s; rq[tid] = ss; __syncthreads();
  for (int st = 64; st > 0; st >>= 1) {
    if (tid < st) { rs[tid] += rs[tid + st]; rq[tid] += rq[tid + st]; }
    __syncthreads();
  }
  float mean = rs[0] * (1.f / DH);
  float var  = rq[0] * (1.f / DH) - mean * mean;
  float inv  = rsqrtf(var + 1e-5f);
  float4 r;
  int c = tid * 4;
  r.x = (vv[0] - mean) * inv * g[c + 0] + be[c + 0];
  r.y = (vv[1] - mean) * inv * g[c + 1] + be[c + 1];
  r.z = (vv[2] - mean) * inv * g[c + 2] + be[c + 2];
  r.w = (vv[3] - mean) * inv * g[c + 3] + be[c + 3];
  o[tid] = r;
}

__global__ __launch_bounds__(256) void k_softmax_rows(float* __restrict__ x, int len) {
  __shared__ float red[256];
  size_t row = blockIdx.x;
  float* p = x + row * (size_t)len;
  int tid = threadIdx.x;
  float m = -3.4e38f;
  for (int i = tid; i < len; i += 256) m = fmaxf(m, p[i]);
  red[tid] = m; __syncthreads();
  for (int st = 128; st > 0; st >>= 1) {
    if (tid < st) red[tid] = fmaxf(red[tid], red[tid + st]);
    __syncthreads();
  }
  m = red[0]; __syncthreads();
  float s = 0.f;
  for (int i = tid; i < len; i += 256) { float e = __expf(p[i] - m); p[i] = e; s += e; }
  red[tid] = s; __syncthreads();
  for (int st = 128; st > 0; st >>= 1) {
    if (tid < st) red[tid] += red[tid + st];
    __syncthreads();
  }
  float inv = 1.f / red[0];
  for (int i = tid; i < len; i += 256) p[i] *= inv;
}

__global__ void k_zero1(float* s) { if (threadIdx.x == 0) s[0] = 0.f; }

// thresh = sigmoid( mean( kv.reshape(b,a,h*d) @ tW + tb ) )
// flat kv[b,h,a,d] maps to column c = (a%8)*64 + d (since a*d = 32*512)
__global__ __launch_bounds__(256) void k_thresh_reduce(const float* __restrict__ kv,
                                                       const float* __restrict__ tW,
                                                       float* __restrict__ scal) {
  __shared__ float red[256];
  const long long total = (long long)BATCH * NH * AG * HD;
  float s = 0.f;
  for (long long i = (long long)blockIdx.x * 256 + threadIdx.x; i < total;
       i += (long long)gridDim.x * 256) {
    int d = (int)(i & 63);
    int a = (int)((i >> 6) & 255);
    s += kv[i] * tW[((a & 7) << 6) + d];
  }
  red[threadIdx.x] = s; __syncthreads();
  for (int st = 128; st > 0; st >>= 1) {
    if (threadIdx.x < st) red[threadIdx.x] += red[threadIdx.x + st];
    __syncthreads();
  }
  if (threadIdx.x == 0) atomicAdd(&scal[0], red[0]);
}

__global__ void k_thresh_final(float* scal, const float* tb) {
  if (threadIdx.x == 0) {
    float x = scal[0] * (1.f / (BATCH * AG)) + tb[0];
    scal[1] = 1.f / (1.f + __expf(-x));
  }
}

// per (b,h,a) row: denoise/mask, then kv2 = softmax(kv*mask + denoise) over d=64
__global__ __launch_bounds__(64) void k_dn_mask_softmax(const float* __restrict__ kv,
                                                        const float* __restrict__ dW,
                                                        const float* __restrict__ db,
                                                        const float* __restrict__ mW,
                                                        const float* __restrict__ mb,
                                                        const float* __restrict__ scal,
                                                        float* __restrict__ kv2) {
  __shared__ float sk[64];
  __shared__ float red[64];
  size_t row = blockIdx.x;
  const float* kr = kv + row * 64;
  int c = threadIdx.x;
  sk[c] = kr[c];
  __syncthreads();
  float ad = db[c], am = mb[c];
  #pragma unroll 8
  for (int k = 0; k < 64; ++k) {
    float kvv = sk[k];
    ad += kvv * dW[k * 64 + c];
    am += kvv * mW[k * 64 + c];
  }
  float den = 1.f / (1.f + __expf(-ad));
  float msk = 1.f / (1.f + __expf(-am));
  float t = sk[c] * ((msk > scal[1]) ? 1.f : 0.f) + den;
  red[c] = t; __syncthreads();
  for (int st = 32; st > 0; st >>= 1) {
    if (c < st) red[c] = fmaxf(red[c], red[c + st]);
    __syncthreads();
  }
  float m = red[0]; __syncthreads();
  float e = __expf(t - m);
  red[c] = e; __syncthreads();
  for (int st = 32; st > 0; st >>= 1) {
    if (c < st) red[c] += red[c + st];
    __syncthreads();
  }
  kv2[row * 64 + c] = e / red[0];
}

// y[b,n,c] = x[b,n,c] + attno[b, c/64, n, c%64]
__global__ __launch_bounds__(256) void k_residual_merge(const float* __restrict__ x,
                                                        const float* __restrict__ attno,
                                                        float* __restrict__ y) {
  const size_t total = (size_t)BATCH * TOK * DH;
  for (size_t i = (size_t)blockIdx.x * blockDim.x + threadIdx.x; i < total;
       i += (size_t)gridDim.x * blockDim.x) {
    size_t b = i / ((size_t)TOK * DH);
    size_t r = i - b * (size_t)TOK * DH;
    size_t n = r / DH;
    int c = (int)(r - n * DH);
    size_t src = ((b * NH + (size_t)(c >> 6)) * TOK + n) * HD + (c & 63);
    y[i] = x[i] + attno[src];
  }
}

__global__ void k_set_cls(float* x, const float* cls) {
  int tid = blockIdx.x * blockDim.x + threadIdx.x;
  if (tid < BATCH * DH) {
    int b = tid >> 9, c = tid & 511;
    x[(size_t)b * TOK * DH + c] = cls[c];
  }
}

__global__ void k_copy_cls(float* dst, const float* src) {
  int tid = blockIdx.x * blockDim.x + threadIdx.x;
  if (tid < BATCH * DH) {
    int b = tid >> 9, c = tid & 511;
    size_t off = (size_t)b * TOK * DH + c;
    dst[off] = src[off];
  }
}

// PPEG: out = x + dwconv7(x) + dwconv5(x) + dwconv3(x)  over the 100x100 plane
__global__ __launch_bounds__(256) void k_ppeg(const float* __restrict__ yin,
                                              float* __restrict__ xout,
                                              const float* __restrict__ w7,
                                              const float* __restrict__ b7,
                                              const float* __restrict__ w5,
                                              const float* __restrict__ b5,
                                              const float* __restrict__ w3,
                                              const float* __restrict__ b3) {
  __shared__ float t[38][38];
  __shared__ float wsm[83];
  int plane = blockIdx.z;
  int b = plane >> 9, c = plane & 511;
  int ti = blockIdx.x, tj = blockIdx.y;
  int tid = threadIdx.x;
  const float* src = yin + ((size_t)b * TOK + 1) * DH + c;
  int gi0 = ti * 32 - 3, gj0 = tj * 32 - 3;
  for (int i = tid; i < 38 * 38; i += 256) {
    int r = i / 38, cc = i - r * 38;
    int gi = gi0 + r, gj = gj0 + cc;
    float v = 0.f;
    if (gi >= 0 && gi < GRID && gj >= 0 && gj < GRID)
      v = src[(size_t)(gi * GRID + gj) * DH];
    t[r][cc] = v;
  }
  if (tid < 49)      wsm[tid] = w7[c * 49 + tid];
  else if (tid < 74) wsm[tid] = w5[c * 25 + (tid - 49)];
  else if (tid < 83) wsm[tid] = w3[c * 9 + (tid - 74)];
  __syncthreads();
  float bias = b7[c] + b5[c] + b3[c];
  for (int q = 0; q < 4; ++q) {
    int p = tid + q * 256;
    int li = p >> 5, lj = p & 31;
    int gi = ti * 32 + li, gj = tj * 32 + lj;
    if (gi < GRID && gj < GRID) {
      float acc = t[li + 3][lj + 3] + bias;
      #pragma unroll
      for (int u = 0; u < 7; ++u)
        #pragma unroll
        for (int v = 0; v < 7; ++v)
          acc += wsm[u * 7 + v] * t[li + u][lj + v];
      #pragma unroll
      for (int u = 0; u < 5; ++u)
        #pragma unroll
        for (int v = 0; v < 5; ++v)
          acc += wsm[49 + u * 5 + v] * t[li + 1 + u][lj + 1 + v];
      #pragma unroll
      for (int u = 0; u < 3; ++u)
        #pragma unroll
        for (int v = 0; v < 3; ++v)
          acc += wsm[74 + u * 3 + v] * t[li + 2 + u][lj + 2 + v];
      xout[((size_t)b * TOK + 1 + gi * GRID + gj) * DH + c] = acc;
    }
  }
}

// final: LN(row0) @ fcW + fcb  ->  d_out (2x2)
__global__ __launch_bounds__(128) void k_final(const float* __restrict__ y,
                                               const float* __restrict__ gf,
                                               const float* __restrict__ bf,
                                               const float* __restrict__ fcW,
                                               const float* __restrict__ fcb,
                                               float* __restrict__ out) {
  __shared__ float r0[128], r1[128];
  int b = blockIdx.x, tid = threadIdx.x;
  const float4* row = (const float4*)(y + (size_t)b * TOK * DH);
  float4 v = row[tid];
  float vv[4] = {v.x, v.y, v.z, v.w};
  float s = vv[0] + vv[1] + vv[2] + vv[3];
  float ss = vv[0]*vv[0] + vv[1]*vv[1] + vv[2]*vv[2] + vv[3]*vv[3];
  r0[tid] = s; r1[tid] = ss; __syncthreads();
  for (int st = 64; st > 0; st >>= 1) {
    if (tid < st) { r0[tid] += r0[tid + st]; r1[tid] += r1[tid + st]; }
    __syncthreads();
  }
  float mean = r0[0] * (1.f / DH);
  float var  = r1[0] * (1.f / DH) - mean * mean;
  float inv  = rsqrtf(var + 1e-5f);
  __syncthreads();
  float p0 = 0.f, p1 = 0.f;
  #pragma unroll
  for (int j = 0; j < 4; ++j) {
    int c = tid * 4 + j;
    float xn = (vv[j] - mean) * inv * gf[c] + bf[c];
    p0 += xn * fcW[c * 2 + 0];
    p1 += xn * fcW[c * 2 + 1];
  }
  r0[tid] = p0; r1[tid] = p1; __syncthreads();
  for (int st = 64; st > 0; st >>= 1) {
    if (tid < st) { r0[tid] += r0[tid + st]; r1[tid] += r1[tid + st]; }
    __syncthreads();
  }
  if (tid == 0) {
    out[b * 2 + 0] = r0[0] + fcb[0];
    out[b * 2 + 1] = r1[0] + fcb[1];
  }
}

// =====================  host orchestration  =====================

static inline void launch_gemm(const GemmP& p, int Z, hipStream_t s) {
  dim3 g((p.M + TM - 1) / TM, (p.N + TN - 1) / TN, Z);
  k_gemm<<<g, dim3(256), 0, s>>>(p);
}

extern "C" void kernel_launch(void* const* d_in, const int* in_sizes, int n_in,
                              void* d_out, int out_size, void* d_ws, size_t ws_size,
                              hipStream_t stream) {
  const float* feats = (const float*)d_in[0];
  const float* W1    = (const float*)d_in[1];
  const float* b1    = (const float*)d_in[2];
  const float* cls   = (const float*)d_in[3];
  const float* g1    = (const float*)d_in[4];
  const float* be1   = (const float*)d_in[5];
  const float* qkv1  = (const float*)d_in[6];
  const float* ag1   = (const float*)d_in[7];
  const float* dW1   = (const float*)d_in[8];
  const float* db1   = (const float*)d_in[9];
  const float* mW1   = (const float*)d_in[10];
  const float* mb1   = (const float*)d_in[11];
  const float* tW1   = (const float*)d_in[12];
  const float* tb1   = (const float*)d_in[13];
  const float* w7    = (const float*)d_in[14];
  const float* b7    = (const float*)d_in[15];
  const float* w5    = (const float*)d_in[16];
  const float* b5    = (const float*)d_in[17];
  const float* w3    = (const float*)d_in[18];
  const float* b3    = (const float*)d_in[19];
  const float* g2    = (const float*)d_in[20];
  const float* be2   = (const float*)d_in[21];
  const float* qkv2  = (const float*)d_in[22];
  const float* ag2   = (const float*)d_in[23];
  const float* dW2   = (const float*)d_in[24];
  const float* db2   = (const float*)d_in[25];
  const float* mW2   = (const float*)d_in[26];
  const float* mb2   = (const float*)d_in[27];
  const float* tW2   = (const float*)d_in[28];
  const float* tb2   = (const float*)d_in[29];
  const float* gf    = (const float*)d_in[30];
  const float* bf    = (const float*)d_in[31];
  const float* fcW   = (const float*)d_in[32];
  const float* fcb   = (const float*)d_in[33];

  // workspace layout (all f32)
  float* w = (float*)d_ws;
  size_t o = 0;
  float* buf_x     = w + o; o += (size_t)BATCH * TOK * DH;
  float* buf_ln    = w + o; o += (size_t)BATCH * TOK * DH;   // doubles as residual output y
  float* buf_qkv   = w + o; o += (size_t)BATCH * TOK * Q3;
  float* buf_qa    = w + o; o += (size_t)BATCH * NH * TOK * AG;
  float* buf_ka    = w + o; o += (size_t)BATCH * NH * AG * TOK;
  float* buf_kv    = w + o; o += (size_t)BATCH * NH * AG * HD;
  float* buf_kv2   = w + o; o += (size_t)BATCH * NH * AG * HD;
  float* buf_attno = w + o; o += (size_t)BATCH * NH * TOK * HD;
  float* scal      = w + o; o += 2;
  (void)ws_size; (void)n_in; (void)in_sizes; (void)out_size;

  auto attn_layer = [&](const float* g, const float* be, const float* qkvW,
                        const float* agent, const float* dW, const float* db,
                        const float* mW, const float* mb, const float* tW,
                        const float* tb, const float* xin, float* yout) {
    // LN
    k_layernorm<<<BATCH * TOK, 128, 0, stream>>>(xin, buf_ln, g, be);
    // qkv = ln @ Wqkv   (no bias in reference)
    GemmP pq{buf_ln, 0, 0, DH,
             qkvW, 0, 0, Q3, 0,
             buf_qkv, 0, 0, Q3,
             nullptr, BATCH * TOK, Q3, DH, 1, 0, 1, 1, 1.f};
    launch_gemm(pq, 1, stream);
    // qa scores = scale * q @ agent^T, then softmax over a
    GemmP pqa{buf_qkv, (long long)TOK * Q3, HD, Q3,
              agent, 0, (long long)AG * HD, HD, 1,
              buf_qa, (long long)NH * TOK * AG, (long long)TOK * AG, AG,
              nullptr, TOK, AG, HD, NH, 0, 1, 1, 0.125f};
    launch_gemm(pqa, BATCH * NH, stream);
    k_softmax_rows<<<BATCH * NH * TOK, 256, 0, stream>>>(buf_qa, AG);
    // ka scores = agent @ k^T, softmax over n
    GemmP pka{agent, 0, (long long)AG * HD, HD,
              buf_qkv + DH, (long long)TOK * Q3, HD, Q3, 1,
              buf_ka, (long long)NH * AG * TOK, (long long)AG * TOK, TOK,
              nullptr, AG, TOK, HD, NH, 0, 1, 1, 1.f};
    launch_gemm(pka, BATCH * NH, stream);
    k_softmax_rows<<<BATCH * NH * AG, 256, 0, stream>>>(buf_ka, TOK);
    // kv = ka @ v   (lda = 10001 -> scalar A staging path)
    GemmP pkv{buf_ka, (long long)NH * AG * TOK, (long long)AG * TOK, TOK,
              buf_qkv + 2 * DH, (long long)TOK * Q3, HD, Q3, 0,
              buf_kv, (long long)NH * AG * HD, (long long)AG * HD, HD,
              nullptr, AG, HD, TOK, NH, 0, 0, 1, 1.f};
    launch_gemm(pkv, BATCH * NH, stream);
    // thresh scalar
    k_zero1<<<1, 32, 0, stream>>>(scal);
    k_thresh_reduce<<<256, 256, 0, stream>>>(buf_kv, tW, scal);
    k_thresh_final<<<1, 32, 0, stream>>>(scal, tb);
    // denoise / mask / softmax over d
    k_dn_mask_softmax<<<BATCH * NH * AG, 64, 0, stream>>>(buf_kv, dW, db, mW, mb,
                                                          scal, buf_kv2);
    // out = qa @ kv2
    GemmP po{buf_qa, (long long)NH * TOK * AG, (long long)TOK * AG, AG,
             buf_kv2, (long long)NH * AG * HD, (long long)AG * HD, HD, 0,
             buf_attno, (long long)NH * TOK * HD, (long long)TOK * HD, HD,
             nullptr, TOK, HD, AG, NH, 0, 1, 1, 1.f};
    launch_gemm(po, BATCH * NH, stream);
    // residual merge (head-interleave back to (b,n,512))
    k_residual_merge<<<2048, 256, 0, stream>>>(xin, buf_attno, yout);
  };

  // 1) h = relu(feats @ W1 + b1) -> buf_x rows 1..10000 per batch
  GemmP p1{feats, (long long)NTOK0 * DI, 0, DI,
           W1, 0, 0, DH, 0,
           buf_x + DH, (long long)TOK * DH, 0, DH,
           b1, NTOK0, DH, DI, 1, 1, 1, 1, 1.f};
  launch_gemm(p1, BATCH, stream);
  k_set_cls<<<4, 256, 0, stream>>>(buf_x, cls);

  // 2) layer 1 attention: y1 = x1 + attn(LN(x1))   (y1 lives in buf_ln)
  attn_layer(g1, be1, qkv1, ag1, dW1, db1, mW1, mb1, tW1, tb1, buf_x, buf_ln);

  // 3) PPEG: buf_ln (y1) -> buf_x
  k_copy_cls<<<4, 256, 0, stream>>>(buf_x, buf_ln);
  k_ppeg<<<dim3(4, 4, BATCH * DH), 256, 0, stream>>>(buf_ln, buf_x,
                                                     w7, b7, w5, b5, w3, b3);

  // 4) layer 2 attention: y2 = x2 + attn(LN(x2))   (y2 lives in buf_ln)
  attn_layer(g2, be2, qkv2, ag2, dW2, db2, mW2, mb2, tW2, tb2, buf_x, buf_ln);

  // 5) final LN(row0) @ fcW + fcb
  k_final<<<BATCH, 128, 0, stream>>>(buf_ln, gf, bf, fcW, fcb, (float*)d_out);
}
